// K2311_2181843386511
// MI455X (gfx1250) — matrix operand order repaired and checked by reading, not compiled, not run on hardware
//
#include <hip/hip_runtime.h>
#include <hip/hip_bf16.h>

// ---- problem constants (match reference) ----
#define NN   8
#define CC   256
#define HW   56
#define LL   3136      // 56*56
#define PLW  62        // padded width (56 + 2*3)
#define PL   3872      // padded plane stride: 62*62=3844 rounded up to 32
#define GG   8
#define CGN  32        // C/G
#define PKK  288       // CG*9
#define SCLEN 9216     // PKK*CGN

typedef __bf16 bf16_t;
typedef __attribute__((ext_vector_type(16))) __bf16 v16bf;
typedef __attribute__((ext_vector_type(8)))  float  v8f;

// ---------- helpers ----------
__device__ __forceinline__ bf16_t f2bf(float f) {
  unsigned u = __builtin_bit_cast(unsigned, f);
  unsigned r = u + 0x7FFFu + ((u >> 16) & 1u);   // round-to-nearest-even
  unsigned short h = (unsigned short)(r >> 16);
  return __builtin_bit_cast(bf16_t, h);
}
__device__ __forceinline__ float bf2f(bf16_t b) {
  unsigned short s = __builtin_bit_cast(unsigned short, b);
  unsigned u = ((unsigned)s) << 16;
  return __builtin_bit_cast(float, u);
}

// async global -> LDS copy (CDNA5 GLOBAL_LOAD_ASYNC_TO_LDS, tracked by ASYNCcnt)
__device__ __forceinline__ void async_ld_b128(unsigned lds_off, const void* g) {
  asm volatile("global_load_async_to_lds_b128 %0, %1, off"
               :: "v"(lds_off), "v"((unsigned long long)(uintptr_t)g)
               : "memory");
}
__device__ __forceinline__ void wait_async0() {
  asm volatile("s_wait_asynccnt 0" ::: "memory");
}
__device__ __forceinline__ void wait_ds0() {
  asm volatile("s_wait_dscnt 0" ::: "memory");
}

// ---------- K0: zero the padded region (halos + NaN guards); ws is 0xAA-poisoned ----------
__global__ __launch_bounds__(256) void k_init(uint4* __restrict__ p, unsigned n16) {
  unsigned i = blockIdx.x * 256u + threadIdx.x;
  const unsigned stride = gridDim.x * 256u;
  const uint4 z = {0u, 0u, 0u, 0u};
  for (; i < n16; i += stride) p[i] = z;
}

// ---------- K1: 13x13 depthwise gather conv via LDS input tile ----------
// one block = one 16x16 output tile; 28x28 zero-filled input tile staged in LDS,
// then 169 unconditional ds_load + fma per output pixel.
__global__ __launch_bounds__(256) void k_gather(const float* __restrict__ x,
                                                const float* __restrict__ w,
                                                const float* __restrict__ bias,
                                                float* __restrict__ xgpadf,
                                                bf16_t* __restrict__ xgpadbf,
                                                bf16_t* __restrict__ sinpadbf) {
  const int plane = blockIdx.x >> 4;          // n*256 + c   (16 tiles per plane)
  const int tile  = blockIdx.x & 15;
  const int ty0   = (tile >> 2) << 4, tx0 = (tile & 3) << 4;
  const int c     = plane & 255;

  __shared__ float lt[28][28];                // input tile + 6-halo (zero-filled)
  __shared__ float lw[169];

  for (int i = threadIdx.x; i < 169; i += 256) lw[i] = w[c * 169 + i];

  const float* xp = x + (size_t)plane * LL;
  for (int i = threadIdx.x; i < 784; i += 256) {
    const int sy = i / 28, sx = i - (i / 28) * 28;
    const int gy = ty0 + sy - 6, gx = tx0 + sx - 6;
    const int ok = ((unsigned)gy < (unsigned)HW) & ((unsigned)gx < (unsigned)HW);
    const float v = xp[ok ? gy * HW + gx : 0];
    lt[sy][sx] = ok ? v : 0.0f;
  }
  __syncthreads();

  const int oy = threadIdx.x >> 4, ox = threadIdx.x & 15;
  const int y = ty0 + oy, xc = tx0 + ox;
  if (y >= HW || xc >= HW) return;

  float acc = bias[c];
  for (int i = 0; i < 13; ++i) {
    #pragma unroll
    for (int j = 0; j < 13; ++j)
      acc = fmaf(lt[oy + i][ox + j], lw[i * 13 + j], acc);
  }
  const size_t o = (size_t)plane * PL + (size_t)(y + 3) * PLW + (xc + 3);
  xgpadf[o]   = acc;
  xgpadbf[o]  = f2bf(acc);
  sinpadbf[o] = f2bf(__sinf(acc * 1.5707963267948966f));
}

// ---------- K2: grouped dilated 3x3 proj conv (from padded xg) -> px (bf16) ----------
__global__ __launch_bounds__(256) void k_proj(const float* __restrict__ xgpadf,
                                              const float* __restrict__ pw,
                                              const float* __restrict__ pb,
                                              bf16_t* __restrict__ pxbf) {
  const int plane = blockIdx.x / 13;          // n*256 + cout
  const int lblk  = blockIdx.x % 13;
  const int cout  = plane & 255;
  const int n     = plane >> 8;
  const int g     = cout >> 5;

  __shared__ float lw[288];
  for (int i = threadIdx.x; i < 288; i += 256) lw[i] = pw[cout * 288 + i];
  __syncthreads();

  const int l = lblk * 256 + threadIdx.x;
  if (l >= LL) return;
  const int y = l / HW, xc = l - y * HW;
  const int ppos = (y + 3) * PLW + (xc + 3);
  const float* base = xgpadf + ((size_t)n * 256 + g * 32) * PL;

  // compile-time tap shifts in padded coords: ((t/3-1)*3)*62 + ((t%3)-1)*3
  constexpr int SH[9] = {-189, -186, -183, -3, 0, 3, 183, 186, 189};

  float acc = pb[cout];
  for (int ic = 0; ic < 32; ++ic) {
    const float* ch = base + (size_t)ic * PL + ppos;
    #pragma unroll
    for (int t = 0; t < 9; ++t)
      acc = fmaf(ch[SH[t]], lw[ic * 9 + t], acc);
  }
  pxbf[(size_t)plane * LL + l] = f2bf(acc);
}

// ---------- K3: attn = (unfold(xg) @ xg^T) / 56 * scale, stored transposed bf16 ----------
// K runs over the PADDED grid (halo B-columns are zero => contribute nothing).
// 192-thread blocks: 6 waves = 6 M-tiles; threads 0..127 async-stage the shared B tile.
__global__ __launch_bounds__(192) void k_attn(const bf16_t* __restrict__ xgpadbf,
                                              const float* __restrict__ scale,
                                              bf16_t* __restrict__ attnT) {
  const int batch = blockIdx.x / 3;           // n*8 + g
  const int bt    = blockIdx.x % 3;
  const int g     = batch & 7;
  const int wave  = threadIdx.x >> 5;
  const int lane  = threadIdx.x & 31;
  const int m     = lane & 15, hi = lane >> 4;
  const int mtile = bt * 6 + wave;            // 0..17

  __shared__ bf16_t tile[2][32][32];          // [buf][channel][k-local]

  // A row (unfold row) for this lane: pure constant shift in padded coords
  const int p  = mtile * 16 + m;              // 0..287
  const int cg = p / 9, kk = p % 9;
  const int shift = (kk / 3 - 1) * 3 * PLW + (kk % 3 - 1) * 3;
  const bf16_t* aShift = xgpadbf + (size_t)(batch * 32 + cg) * PL + shift;
  const bf16_t* bbase  = xgpadbf + (size_t)batch * 32 * PL;

  // staging assignment: threads 0..127 -> row = tid/4, 16B quarter = tid%4
  const int srow = threadIdx.x >> 2, sq = threadIdx.x & 3;
  const bf16_t* ssrc = bbase + (size_t)srow * PL + sq * 8;

  v8f acc0 = {}; v8f acc1 = {};

  if (threadIdx.x < 128)
    async_ld_b128((unsigned)(uintptr_t)&tile[0][srow][sq * 8], ssrc);

  for (int it = 0; it < 121; ++it) {          // 121*32 = 3872 = PL
    const int k0  = it * 32;
    const int buf = it & 1;

    wait_ds0();                               // my reads of tile[buf^1] done (WAR)
    wait_async0();                            // my staged writes landed (RAW)
    __syncthreads();

    if (it + 1 < 121 && threadIdx.x < 128) {
      async_ld_b128((unsigned)(uintptr_t)&tile[buf ^ 1][srow][sq * 8], ssrc + k0 + 32);
      __builtin_prefetch(ssrc + k0 + 256, 0, 1);
    }

    // A fragment: contiguous unconditional loads (merged to b128 by the compiler)
    v16bf a;
    {
      const bf16_t* p0 = aShift + k0 + hi * 16;
      #pragma unroll
      for (int j = 0; j < 8; ++j) a[j] = p0[j];
      const bf16_t* p1 = p0 + 8;
      #pragma unroll
      for (int j = 0; j < 8; ++j) a[8 + j] = p1[j];
    }

    // B fragments from LDS (lane n holds K = hi*16 + 0..15 of column n)
    const v16bf b0 = *(const v16bf*)&tile[buf][m][hi * 16];
    const v16bf b1 = *(const v16bf*)&tile[buf][16 + m][hi * 16];

    acc0 = __builtin_amdgcn_wmma_f32_16x16x32_bf16(false, a, false, b0, (short)0, acc0, false, false);
    acc1 = __builtin_amdgcn_wmma_f32_16x16x32_bf16(false, a, false, b1, (short)0, acc1, false, false);
  }

  // scale and store transposed: attnT[batch][c][p]
  const float* sc = scale + g * SCLEN;
  #pragma unroll
  for (int r = 0; r < 8; ++r) {
    const int prow = mtile * 16 + r + 8 * hi;
    const float v0 = acc0[r] * (1.0f / 56.0f) * sc[prow * 32 + m];
    const float v1 = acc1[r] * (1.0f / 56.0f) * sc[prow * 32 + 16 + m];
    attnT[(size_t)(batch * 32 + m) * PKK + prow]      = f2bf(v0);
    attnT[(size_t)(batch * 32 + 16 + m) * PKK + prow] = f2bf(v1);
  }
}

// ---------- K4: out = attn^T @ max(px, sin_unfold), / sqrt(288) ----------
// 128-thread blocks: 4 waves = 4 L-tiles; every thread async-stages exactly one 16B chunk.
// Lane-invariant gather offsets precomputed in an LDS table.
__global__ __launch_bounds__(128) void k_out(const bf16_t* __restrict__ attnT,
                                             const bf16_t* __restrict__ pxbf,
                                             const bf16_t* __restrict__ sinpadbf,
                                             float* __restrict__ out) {
  const int batch = blockIdx.x / 49;
  const int ltb   = blockIdx.x % 49;
  const int wave  = threadIdx.x >> 5;
  const int lane  = threadIdx.x & 31;
  const int m     = lane & 15, hi = lane >> 4;
  const int lt    = ltb * 4 + wave;            // 0..195
  const int l     = lt * 16 + m;               // B column for this lane
  const int y     = l / HW, xc = l - y * HW;
  const int ppos  = (y + 3) * PLW + (xc + 3);

  __shared__ bf16_t tA[2][32][32];             // [buf][row c][k-local]
  __shared__ int2   offT[PKK];                 // {cg*PL + sin-shift, cg*LL}

  const bf16_t* aBase = attnT    + (size_t)batch * 32 * PKK;
  const bf16_t* pxb   = pxbf     + (size_t)batch * 32 * LL;
  const bf16_t* sinb  = sinpadbf + (size_t)batch * 32 * PL;

  for (int i = threadIdx.x; i < PKK; i += 128) {
    const int cg = i / 9, t = i - (i / 9) * 9;
    const int sh = (t / 3) * (3 * PLW) + (t % 3) * 3 - (3 * PLW + 3);
    offT[i] = make_int2(cg * PL + sh, cg * LL);
  }

  const int srow = threadIdx.x >> 2, sq = threadIdx.x & 3;  // 32 rows x 4 quarters
  const bf16_t* ssrc = aBase + (size_t)srow * PKK + sq * 8;

  v8f acc0 = {}; v8f acc1 = {};

  async_ld_b128((unsigned)(uintptr_t)&tA[0][srow][sq * 8], ssrc);

  #pragma unroll 1
  for (int it = 0; it < 9; ++it) {
    const int k0  = it * 32;
    const int buf = it & 1;

    wait_ds0();
    wait_async0();
    __syncthreads();                           // also covers the offT table build (it 0)

    if (it + 1 < 9)
      async_ld_b128((unsigned)(uintptr_t)&tA[buf ^ 1][srow][sq * 8], ssrc + k0 + 32);

    // A fragments from LDS (rows m and 16+m), two aligned 16B chunks each
    union { v16bf v; uint4 q[2]; } a0, a1;
    a0.q[0] = *(const uint4*)&tA[buf][m][hi * 8];
    a0.q[1] = *(const uint4*)&tA[buf][m][16 + hi * 8];
    a1.q[0] = *(const uint4*)&tA[buf][16 + m][hi * 8];
    a1.q[1] = *(const uint4*)&tA[buf][16 + m][16 + hi * 8];

    // B fragment: fused = max(px[cg,l], sin_unf[cg,kk,l]); offsets from LDS table
    v16bf b;
    #pragma unroll
    for (int j = 0; j < 16; ++j) {
      const int kg = k0 + hi * 8 + j + (j >= 8 ? 8 : 0);  // 0..287
      const int2 o = offT[kg];
      const float sv = bf2f(sinb[o.x + ppos]);
      const float pv = bf2f(pxb[o.y + l]);
      b[j] = f2bf(fmaxf(pv, sv));
    }

    acc0 = __builtin_amdgcn_wmma_f32_16x16x32_bf16(false, a0.v, false, b, (short)0, acc0, false, false);
    acc1 = __builtin_amdgcn_wmma_f32_16x16x32_bf16(false, a1.v, false, b, (short)0, acc1, false, false);
  }

  const float s = 0.05892556509887896f;        // 1/sqrt(288)
  #pragma unroll
  for (int r = 0; r < 8; ++r) {
    const int cg0 = r + 8 * hi;
    out[(size_t)(batch * 32 + cg0) * LL + l]      = acc0[r] * s;
    out[(size_t)(batch * 32 + 16 + cg0) * LL + l] = acc1[r] * s;
  }
}

// ---------- host ----------
extern "C" void kernel_launch(void* const* d_in, const int* in_sizes, int n_in,
                              void* d_out, int out_size, void* d_ws, size_t ws_size,
                              hipStream_t stream) {
  const float* x        = (const float*)d_in[0];
  const float* gather_w = (const float*)d_in[1];
  const float* gather_b = (const float*)d_in[2];
  const float* proj_w   = (const float*)d_in[3];
  const float* proj_b   = (const float*)d_in[4];
  const float* scale    = (const float*)d_in[5];
  float* out = (float*)d_out;

  // workspace layout (all regions 16B aligned)
  char* w = (char*)d_ws;
  const size_t padF32 = (size_t)NN * CC * PL * sizeof(float);   // 31,719,424
  const size_t padBF  = (size_t)NN * CC * PL * sizeof(bf16_t);  // 15,859,712
  float*  xgpadf   = (float*)w;    w += padF32;
  /* guard (zeroed) */             w += 1024;                   // 512 bf16 zeros
  bf16_t* xgpadbf  = (bf16_t*)w;   w += padBF;
  /* guard (zeroed) */             w += 1024;                   // 512 bf16 zeros
  bf16_t* sinpadbf = (bf16_t*)w;   w += padBF;
  bf16_t* pxbf     = (bf16_t*)w;   w += (size_t)NN * CC * LL * sizeof(bf16_t);
  bf16_t* attnT    = (bf16_t*)w;                                // 64*32*288*2 bytes

  // zero the whole padded span (halos, guards) in one grid-stride memset
  const size_t zeroBytes = padF32 + 1024 + padBF + 1024 + padBF;
  k_init<<<4096, 256, 0, stream>>>((uint4*)d_ws, (unsigned)(zeroBytes / 16));

  dim3 gridTile(NN * CC * 16);                                  // 16 16x16 tiles per plane
  k_gather<<<gridTile, 256, 0, stream>>>(x, gather_w, gather_b, xgpadf, xgpadbf, sinpadbf);

  dim3 gridConv(NN * CC * 13);                                  // 13 blocks per real plane
  k_proj<<<gridConv, 256, 0, stream>>>(xgpadf, proj_w, proj_b, pxbf);

  dim3 gridAttn(64 * 3);                                        // 6 M-tiles per block
  k_attn<<<gridAttn, 192, 0, stream>>>(xgpadbf, scale, attnT);

  dim3 gridOut(64 * 49);                                        // 4 L-tiles per block
  k_out<<<gridOut, 128, 0, stream>>>(attnT, pxbf, sinpadbf, out);
}
